// StructureModuleXYZ_39711267619482
// MI455X (gfx1250) — matrix operand order repaired and checked by reading, not compiled, not run on hardware
//
#include <hip/hip_runtime.h>
#include <math.h>

// ---------------- constants ----------------
#define NRES 512
#define CS   384
#define CZ   128
#define HNUM 12
#define CIPA 16
#define PQK  4
#define PV   8
#define NBLK 8
#define CRES 128
#define NA   7
#define NATOM 23
#define GNUM 9
#define TSF  10.0f
#define EPSF 1e-8f
#define INFF 100000.0f
#define CATW 2112   // H*(CZ + CIPA + PV*4)

#define GEMM_RELU_IN 1
#define GEMM_ACCUM   2

typedef __attribute__((ext_vector_type(16))) _Float16 v16h;
typedef __attribute__((ext_vector_type(8)))  float    v8f;

// ---------------- wave helpers ----------------
__device__ inline float wred(float v) {
  #pragma unroll
  for (int o = 16; o > 0; o >>= 1) v += __shfl_xor(v, o, 32);
  return v;
}

// ---------------- WMMA fragment loaders (16-bit, wave32, per ISA 7.12.2) -----
// A tile: 16 (M) x 32 (K) f16, row-major in LDS, row stride ldm.
// lane 0-15: M=lane, K in {0..7,16..23}; lane 16-31: M=lane-16, K in {8..15,24..31}
__device__ inline v16h load_a_frag(const _Float16* lds, int ldm, int lane) {
  int r = lane & 15, kb = (lane >> 4) * 8;
  v16h a;
  #pragma unroll
  for (int e = 0; e < 8; ++e) {
    a[e]     = lds[r * ldm + kb + e];
    a[e + 8] = lds[r * ldm + kb + 16 + e];
  }
  return a;
}
// B tile: 32 (K) x 16 (N) f16, row-major in LDS, row stride ldn.
// lane 0-15: N=lane, K in {0..7,16..23}; lane 16-31: N=lane-16, K in {8..15,24..31}
__device__ inline v16h load_b_frag(const _Float16* lds, int ldn, int lane) {
  int n = lane & 15, kb = (lane >> 4) * 8;
  v16h b;
  #pragma unroll
  for (int e = 0; e < 8; ++e) {
    b[e]     = lds[(kb + e) * ldn + n];
    b[e + 8] = lds[(kb + 16 + e) * ldn + n];
  }
  return b;
}

// ---------------- elementwise ----------------
__global__ void k_zero(float* p, int n) {
  int i = blockIdx.x * blockDim.x + threadIdx.x;
  if (i < n) p[i] = 0.f;
}
__global__ void k_copy(const float* __restrict__ a, float* __restrict__ b, int n) {
  int i = blockIdx.x * blockDim.x + threadIdx.x;
  if (i < n) b[i] = a[i];
}

// ---------------- LayerNorm (one block per row) ----------------
__global__ void k_ln(const float* __restrict__ x, const float* __restrict__ g,
                     const float* __restrict__ b, float* __restrict__ out, int cols) {
  int row = blockIdx.x, tid = threadIdx.x;
  __shared__ float red[128];
  const float* xr = x + (size_t)row * cols;
  float s = 0.f, s2 = 0.f;
  for (int c = tid; c < cols; c += 128) { float v = xr[c]; s += v; s2 += v * v; }
  red[tid] = s; __syncthreads();
  for (int o = 64; o > 0; o >>= 1) { if (tid < o) red[tid] += red[tid + o]; __syncthreads(); }
  float mean = red[0] / cols; __syncthreads();
  red[tid] = s2; __syncthreads();
  for (int o = 64; o > 0; o >>= 1) { if (tid < o) red[tid] += red[tid + o]; __syncthreads(); }
  float var = red[0] / cols - mean * mean;
  float rs = rsqrtf(var + 1e-5f);
  float* orow = out + (size_t)row * cols;
  for (int c = tid; c < cols; c += 128) orow[c] = (xr[c] - mean) * rs * g[c] + b[c];
}

// ---- generic WMMA GEMM: out[M,Nc] (op)= act(in[M,K]) @ W[K,Nc] + bias -------
// 64x128 block tile, 8 waves; each wave: 4 M-subtiles x one 16-col slice.
__global__ __launch_bounds__(256)
void k_gemm(const float* __restrict__ in, int ldi, const float* __restrict__ W,
            const float* __restrict__ bias, float* __restrict__ out, int ldo,
            int M, int K, int Nc, int flags) {
  __shared__ _Float16 As[64][40];
  __shared__ _Float16 Bs[32][136];
  const int tid = threadIdx.x, lane = tid & 31, wv = tid >> 5;
  const int m0 = blockIdx.x * 64, n0 = blockIdx.y * 128;
  v8f acc[4] = {};
  for (int kk = 0; kk < K; kk += 32) {
    for (int idx = tid; idx < 64 * 32; idx += 256) {
      int r = idx >> 5, c = idx & 31;
      int gr = m0 + r, gc = kk + c;
      float v = (gr < M && gc < K) ? in[(size_t)gr * ldi + gc] : 0.f;
      if (flags & GEMM_RELU_IN) v = fmaxf(v, 0.f);
      As[r][c] = (_Float16)v;
    }
    for (int idx = tid; idx < 32 * 128; idx += 256) {
      int r = idx >> 7, c = idx & 127;
      int gk = kk + r, gn = n0 + c;
      Bs[r][c] = (_Float16)((gk < K && gn < Nc) ? W[(size_t)gk * Nc + gn] : 0.f);
    }
    if (kk + 32 < K) __builtin_prefetch(&W[(size_t)(kk + 32) * Nc + n0 + lane], 0, 1);
    __syncthreads();
    v16h b = load_b_frag(&Bs[0][wv * 16], 136, lane);
    #pragma unroll
    for (int mt = 0; mt < 4; ++mt) {
      v16h a = load_a_frag(&As[mt * 16][0], 40, lane);
      acc[mt] = __builtin_amdgcn_wmma_f32_16x16x32_f16(false, a, false, b, (short)0,
                                                       acc[mt], false, false);
    }
    __syncthreads();
  }
  int col = n0 + wv * 16 + (lane & 15);
  if (col < Nc) {
    float bv = bias ? bias[col] : 0.f;
    #pragma unroll
    for (int mt = 0; mt < 4; ++mt) {
      #pragma unroll
      for (int r = 0; r < 8; ++r) {
        int row = m0 + mt * 16 + r + 8 * (lane >> 4);
        if (row < M) {
          float v = acc[mt][r] + bv;
          float* o = out + (size_t)row * ldo + col;
          if (flags & GEMM_ACCUM) *o += v; else *o = v;
        }
      }
    }
  }
}

// ---------------- pair bias: b[i,j,h] = LN(z[i,j,:]) @ Wb + bb  (wave per row) --
__global__ void k_pair_bias(const float* __restrict__ z, const float* __restrict__ g,
                            const float* __restrict__ b, const float* __restrict__ Wb,
                            const float* __restrict__ bb, float* __restrict__ outb) {
  int wv = threadIdx.x >> 5, lane = threadIdx.x & 31;
  long row = (long)blockIdx.x * 8 + wv;   // 0 .. NRES*NRES-1
  const float* zr = z + (size_t)row * CZ;
  int c0 = lane * 4;
  float v[4], s = 0.f, s2 = 0.f;
  #pragma unroll
  for (int e = 0; e < 4; ++e) { v[e] = zr[c0 + e]; s += v[e]; s2 += v[e] * v[e]; }
  s = wred(s); s2 = wred(s2);
  float mean = s * (1.f / CZ);
  float rs = rsqrtf(s2 * (1.f / CZ) - mean * mean + 1e-5f);
  float zn[4];
  #pragma unroll
  for (int e = 0; e < 4; ++e) zn[e] = (v[e] - mean) * rs * g[c0 + e] + b[c0 + e];
  #pragma unroll
  for (int h = 0; h < HNUM; ++h) {
    float p = 0.f;
    #pragma unroll
    for (int e = 0; e < 4; ++e) p += zn[e] * Wb[(size_t)(c0 + e) * HNUM + h];
    p = wred(p);
    if (lane == 0) outb[(size_t)row * HNUM + h] = p + bb[h];
  }
}

// ---------------- rigid frame from 3 points ----------------
__device__ inline void frame3(const float* p0, const float* p1, const float* p2,
                              float* R, float* t) {
  float e0[3], e1[3];
  #pragma unroll
  for (int d = 0; d < 3; ++d) e0[d] = p1[d] - p0[d];
  float n0 = rsqrtf(e0[0]*e0[0] + e0[1]*e0[1] + e0[2]*e0[2] + EPSF);
  #pragma unroll
  for (int d = 0; d < 3; ++d) e0[d] *= n0;
  #pragma unroll
  for (int d = 0; d < 3; ++d) e1[d] = p2[d] - p1[d];
  float dp = e1[0]*e0[0] + e1[1]*e0[1] + e1[2]*e0[2];
  #pragma unroll
  for (int d = 0; d < 3; ++d) e1[d] -= dp * e0[d];
  float n1 = rsqrtf(e1[0]*e1[0] + e1[1]*e1[1] + e1[2]*e1[2] + EPSF);
  #pragma unroll
  for (int d = 0; d < 3; ++d) e1[d] *= n1;
  float e2[3] = { e0[1]*e1[2] - e0[2]*e1[1],
                  e0[2]*e1[0] - e0[0]*e1[2],
                  e0[0]*e1[1] - e0[1]*e1[0] };
  #pragma unroll
  for (int d = 0; d < 3; ++d) { R[d*3+0] = e0[d]; R[d*3+1] = e1[d]; R[d*3+2] = e2[d]; t[d] = p1[d]; }
}

__global__ void k_frames(const float* __restrict__ xyz, float* __restrict__ R,
                         float* __restrict__ t) {
  int n = blockIdx.x * blockDim.x + threadIdx.x;
  if (n >= NRES) return;
  const float* a = xyz + n * 24;
  frame3(a, a + 3, a + 6, R + n * 9, t + n * 3);
}

__global__ void k_apply(const float* __restrict__ pts, const float* __restrict__ R,
                        const float* __restrict__ t, float* __restrict__ out, int P) {
  int idx = blockIdx.x * blockDim.x + threadIdx.x;
  if (idx >= NRES * P) return;
  int n = idx / P, p = idx % P;
  const float* Rp = R + n * 9; const float* tp = t + n * 3;
  const float* v = pts + (size_t)n * P * 3 + p * 3;
  float* o = out + (size_t)n * P * 3 + p * 3;
  #pragma unroll
  for (int d = 0; d < 3; ++d)
    o[d] = Rp[d*3+0]*v[0] + Rp[d*3+1]*v[1] + Rp[d*3+2]*v[2] + tp[d];
}

// ============ attention stage 1: one block per (i-tile, head) ================
// Fused-K WMMA logits: A row il = [q*s1 (16) | q_pts*hw (12) | 0 (4)]
//                      B col j  = [k (16)    | k_pts (12)    | 0 (4)]
// logit = wmma + s2*bias - 0.5*hw*(q2+k2) + maskterm; softmax over j;
// then o = P@V and o_pt = P@V_pts, both WMMA (K = 512 over j).
__global__ __launch_bounds__(256)
void k_attn1(const float* __restrict__ q, const float* __restrict__ kv,
             const float* __restrict__ qpa, const float* __restrict__ kvpa,
             const float* __restrict__ biasb, const float* __restrict__ mask,
             const float* __restrict__ gamma, _Float16* __restrict__ probsb,
             float* __restrict__ cat, float* __restrict__ optraw) {
  __shared__ _Float16 af[16][36];
  __shared__ float    lgt[16][520];
  __shared__ _Float16 pft[16][520];
  __shared__ _Float16 kb[8][32][24];
  __shared__ _Float16 vb[8][32][56];
  __shared__ float    part[8][16][48];
  __shared__ float    q2s[16], mrow[16];
  __shared__ float    k2s[NRES], mcs[NRES];

  const int it = blockIdx.x, h = blockIdx.y;
  const int i0 = it * 16;
  const int tid = threadIdx.x, lane = tid & 31, wv = tid >> 5;
  const float s1  = 0.1443375673f;   // sqrt(1/(3*CIPA))
  const float s2c = 0.5773502692f;   // sqrt(1/3)
  const float hw  = logf(1.f + __expf(gamma[h])) * 0.1360827635f; // sqrt(1/54)

  // stage fused-Q A tile
  for (int idx = tid; idx < 16 * 32; idx += 256) {
    int row = idx >> 5, col = idx & 31;
    int i = i0 + row;
    float v = 0.f;
    if (col < 16)       v = q[((size_t)i * HNUM + h) * CIPA + col] * s1;
    else if (col < 28)  v = qpa[(size_t)i * 144 + h * 12 + (col - 16)] * hw;
    af[row][col] = (_Float16)v;
  }
  if (tid < 16) {
    int i = i0 + tid;
    float s = 0.f;
    #pragma unroll
    for (int pd = 0; pd < 12; ++pd) {
      float v = qpa[(size_t)i * 144 + h * 12 + pd];
      s += v * v;
    }
    q2s[tid] = s;
    mrow[tid] = mask[i];
  }
  __syncthreads();

  v16h aq = load_a_frag(&af[0][0], 36, lane);

  // logits: 32 j-tiles of 16, wave-local staging (same-wave DS ops are in-order)
  for (int jt = wv; jt < 32; jt += 8) {
    int j0 = jt * 16;
    int jl = lane & 15, half = lane >> 4;
    int j = j0 + jl;
    if (half == 0) {
      #pragma unroll
      for (int c = 0; c < 16; ++c)
        kb[wv][c][jl] = (_Float16)kv[((size_t)j * HNUM + h) * 2 * CIPA + c];
      mcs[j] = mask[j];
    } else {
      float s = 0.f;
      #pragma unroll
      for (int pd = 0; pd < 12; ++pd) {
        float v = kvpa[(size_t)j * 432 + h * 36 + pd];
        kb[wv][16 + pd][jl] = (_Float16)v;
        s += v * v;
      }
      #pragma unroll
      for (int r2 = 28; r2 < 32; ++r2) kb[wv][r2][jl] = (_Float16)0.f;
      k2s[j] = s;
    }
    v16h b = load_b_frag(&kb[wv][0][0], 24, lane);
    v8f acc = {};
    acc = __builtin_amdgcn_wmma_f32_16x16x32_f16(false, aq, false, b, (short)0, acc,
                                                 false, false);
    int jc = j0 + (lane & 15);
    #pragma unroll
    for (int r = 0; r < 8; ++r) {
      int il = r + 8 * (lane >> 4);
      lgt[il][jc] = acc[r]
                  + s2c * biasb[((size_t)(i0 + il) * NRES + jc) * HNUM + h]
                  - 0.5f * hw * (q2s[il] + k2s[jc])
                  + (mrow[il] * mcs[jc] - 1.f) * INFF;
    }
  }
  __syncthreads();

  // softmax per i-row (2 rows per wave)
  for (int il = wv; il < 16; il += 8) {
    float mx = -3.4e38f;
    for (int j = lane; j < NRES; j += 32) mx = fmaxf(mx, lgt[il][j]);
    #pragma unroll
    for (int o = 16; o > 0; o >>= 1) mx = fmaxf(mx, __shfl_xor(mx, o, 32));
    float sm = 0.f;
    for (int j = lane; j < NRES; j += 32) sm += __expf(lgt[il][j] - mx);
    sm = wred(sm);
    float inv = 1.f / sm;
    _Float16* prow = probsb + ((size_t)h * NRES + (i0 + il)) * NRES;
    for (int j = lane; j < NRES; j += 32) {
      float p = __expf(lgt[il][j] - mx) * inv;
      _Float16 ph = (_Float16)p;
      pft[il][j] = ph;
      prow[j] = ph;
    }
  }
  __syncthreads();

  // o (16 cols) + o_pt (24 cols) via WMMA; each wave covers 64 of K=512,
  // partials reduced through LDS.
  v8f acc3[3] = {};
  for (int kc = wv * 64; kc < wv * 64 + 64; kc += 32) {
    int j = kc + lane;           // lane stages one whole j-row of B
    #pragma unroll
    for (int c = 0; c < 16; ++c)
      vb[wv][lane][c] = (_Float16)kv[((size_t)j * HNUM + h) * 2 * CIPA + CIPA + c];
    #pragma unroll
    for (int pd = 0; pd < 24; ++pd)
      vb[wv][lane][16 + pd] = (_Float16)kvpa[(size_t)j * 432 + h * 36 + 12 + pd];
    #pragma unroll
    for (int c = 40; c < 48; ++c) vb[wv][lane][c] = (_Float16)0.f;
    v16h ap = load_a_frag(&pft[0][kc], 520, lane);
    #pragma unroll
    for (int t = 0; t < 3; ++t) {
      v16h bf = load_b_frag(&vb[wv][0][t * 16], 56, lane);
      acc3[t] = __builtin_amdgcn_wmma_f32_16x16x32_f16(false, ap, false, bf, (short)0,
                                                       acc3[t], false, false);
    }
  }
  #pragma unroll
  for (int t = 0; t < 3; ++t)
    #pragma unroll
    for (int r = 0; r < 8; ++r)
      part[wv][r + 8 * (lane >> 4)][t * 16 + (lane & 15)] = acc3[t][r];
  __syncthreads();
  for (int idx = tid; idx < 16 * 40; idx += 256) {
    int il = idx / 40, c = idx % 40;
    float s = 0.f;
    #pragma unroll
    for (int w = 0; w < 8; ++w) s += part[w][il][c];
    int i = i0 + il;
    if (c < 16) cat[(size_t)i * CATW + h * CIPA + c] = s;
    else        optraw[(size_t)i * HNUM * PV * 3 + h * 24 + (c - 16)] = s;
  }
}

// ============ attention stage 2: o_pair per residue (heads x C_Z WMMA) =======
__global__ __launch_bounds__(256)
void k_pair(const _Float16* __restrict__ probsb, const float* __restrict__ z,
            const float* __restrict__ zg, const float* __restrict__ zb,
            float* __restrict__ cat) {
  __shared__ _Float16 pf[16][520];
  __shared__ _Float16 zt[32][136];
  const int i = blockIdx.x, tid = threadIdx.x;
  const int lane = tid & 31, wv = tid >> 5;
  for (int idx = tid; idx < HNUM * NRES; idx += 256) {
    int hh = idx >> 9, j = idx & 511;
    pf[hh][j] = probsb[((size_t)hh * NRES + i) * NRES + j];
  }
  for (int idx = tid; idx < 4 * NRES; idx += 256)
    pf[12 + (idx >> 9)][idx & 511] = (_Float16)0.f;
  __syncthreads();

  v8f acc = {};
  const int c0 = wv * 16;                 // 8 waves * 16 = 128 = CZ
  for (int kc = 0; kc < NRES; kc += 32) {
    for (int rr = 0; rr < 4; ++rr) {
      int jl = wv * 4 + rr;
      const float* zr = z + ((size_t)i * NRES + (kc + jl)) * CZ;
      int cc = lane * 4;
      float v0 = zr[cc], v1 = zr[cc+1], v2 = zr[cc+2], v3 = zr[cc+3];
      float s = wred(v0 + v1 + v2 + v3);
      float sq = wred(v0*v0 + v1*v1 + v2*v2 + v3*v3);
      float mean = s * (1.f / CZ);
      float rs = rsqrtf(sq * (1.f / CZ) - mean * mean + 1e-5f);
      zt[jl][cc]   = (_Float16)((v0 - mean) * rs * zg[cc]   + zb[cc]);
      zt[jl][cc+1] = (_Float16)((v1 - mean) * rs * zg[cc+1] + zb[cc+1]);
      zt[jl][cc+2] = (_Float16)((v2 - mean) * rs * zg[cc+2] + zb[cc+2]);
      zt[jl][cc+3] = (_Float16)((v3 - mean) * rs * zg[cc+3] + zb[cc+3]);
    }
    __syncthreads();
    v16h a = load_a_frag(&pf[0][kc], 520, lane);
    v16h b = load_b_frag(&zt[0][c0], 136, lane);
    acc = __builtin_amdgcn_wmma_f32_16x16x32_f16(false, a, false, b, (short)0, acc,
                                                 false, false);
    __syncthreads();
  }
  int col = c0 + (lane & 15);
  #pragma unroll
  for (int r = 0; r < 8; ++r) {
    int h = r + 8 * (lane >> 4);
    if (h < HNUM) cat[(size_t)i * CATW + 576 + h * CZ + col] = acc[r];
  }
}

// ---------------- inv-apply o_pt, norms, fill cat ----------------
__global__ void k_finalize(const float* __restrict__ optraw, const float* __restrict__ R,
                           const float* __restrict__ t, float* __restrict__ cat) {
  int idx = blockIdx.x * blockDim.x + threadIdx.x;     // NRES * 96
  if (idx >= NRES * HNUM * PV) return;
  int n = idx / (HNUM * PV), hp = idx % (HNUM * PV);
  const float* p = optraw + (size_t)n * HNUM * PV * 3 + hp * 3;
  const float* Rp = R + n * 9; const float* tp = t + n * 3;
  float v[3] = { p[0]-tp[0], p[1]-tp[1], p[2]-tp[2] };
  float o[3];
  #pragma unroll
  for (int d = 0; d < 3; ++d)
    o[d] = Rp[0*3+d]*v[0] + Rp[1*3+d]*v[1] + Rp[2*3+d]*v[2];   // R^T (p - t)
  float* cr = cat + (size_t)n * CATW + HNUM * CIPA;
  cr[hp] = o[0]; cr[96 + hp] = o[1]; cr[192 + hp] = o[2];
  cr[288 + hp] = sqrtf(o[0]*o[0] + o[1]*o[1] + o[2]*o[2] + EPSF);
}

// ---------------- angle normalize ----------------
__global__ void k_angles(const float* __restrict__ un, float* __restrict__ angbuf,
                         float* __restrict__ out_un, float* __restrict__ out_ang) {
  int idx = blockIdx.x * blockDim.x + threadIdx.x;    // NRES*NA
  if (idx >= NRES * NA) return;
  float x = un[idx*2], y = un[idx*2+1];
  float nrm = sqrtf(fmaxf(x*x + y*y, EPSF));
  float ax = x / nrm, ay = y / nrm;
  angbuf[idx*2] = ax; angbuf[idx*2+1] = ay;
  out_un[idx*2] = x; out_un[idx*2+1] = y;
  out_ang[idx*2] = ax; out_ang[idx*2+1] = ay;
}

// ---------------- torsion frames + atom placement (thread per residue) -------
__device__ inline void mm3(const float* A, const float* B, float* C) {
  #pragma unroll
  for (int r = 0; r < 3; ++r)
    #pragma unroll
    for (int c = 0; c < 3; ++c)
      C[r*3+c] = A[r*3+0]*B[0*3+c] + A[r*3+1]*B[1*3+c] + A[r*3+2]*B[2*3+c];
}
__device__ inline void mv3(const float* A, const float* v, float* o) {
  #pragma unroll
  for (int r = 0; r < 3; ++r) o[r] = A[r*3+0]*v[0] + A[r*3+1]*v[1] + A[r*3+2]*v[2];
}

__global__ void k_atoms(const float* __restrict__ xyz, const int* __restrict__ butype,
                        const float* __restrict__ ang, const float* __restrict__ dframes,
                        const int* __restrict__ gidx, const float* __restrict__ lit,
                        const float* __restrict__ amask, float* __restrict__ out_pos) {
  int n = blockIdx.x * blockDim.x + threadIdx.x;
  if (n >= NRES) return;
  bool prot = (n < NRES / 2);                       // protein_pos = [0,N/2)
  const float* X = xyz + n * 24;
  float Rb[2][9], tb[2][3];
  for (int f = 0; f < 2; ++f) { for (int e = 0; e < 9; ++e) Rb[f][e] = 0.f;
                                for (int e = 0; e < 3; ++e) tb[f][e] = 0.f; }
  if (prot) {
    frame3(X + 0, X + 3, X + 6, Rb[0], tb[0]);
  } else {
    float neg3[3] = { -X[9], -X[10], -X[11] };
    frame3(neg3, X + 12, X + 15, Rb[0], tb[0]);
    frame3(neg3, X + 18, X + 21, Rb[1], tb[1]);
  }
  for (int f = 0; f < 2; ++f) for (int e = 0; e < 3; ++e) tb[f][e] *= TSF;

  int bu = butype[n];
  const float* df = dframes + (size_t)bu * GNUM * 16;
  const float* an = ang + n * NA * 2;
  float gR[GNUM][9], gt[GNUM][3];

  // RNA path: 2 identity pads + 7 angles, sel = [0,1,0,...,0]
  for (int g = 0; g < GNUM; ++g) {
    float s = (g < 2) ? 0.f : an[(g-2)*2];
    float c = (g < 2) ? 1.f : an[(g-2)*2+1];
    float rot[9] = {1,0,0, 0,c,-s, 0,s,c};
    float dR[9], dt[3];
    for (int r = 0; r < 3; ++r) { for (int cc = 0; cc < 3; ++cc) dR[r*3+cc] = df[g*16 + r*4 + cc];
                                  dt[r] = df[g*16 + r*4 + 3]; }
    float aR[9]; mm3(dR, rot, aR);
    int sel = (g == 1) ? 1 : 0;
    float tmp[9]; mm3(Rb[sel], aR, tmp);
    float tv[3];  mv3(Rb[sel], dt, tv);
    for (int e = 0; e < 9; ++e) gR[g][e] = tmp[e];
    for (int e = 0; e < 3; ++e) gt[g][e] = tv[e] + tb[sel][e];
  }
  if (prot) {
    float aR[8][9], at[8][3];
    for (int g = 0; g < 8; ++g) {
      float s = (g < 1) ? 0.f : an[(g-1)*2];
      float c = (g < 1) ? 1.f : an[(g-1)*2+1];
      float rot[9] = {1,0,0, 0,c,-s, 0,s,c};
      float dR[9];
      for (int r = 0; r < 3; ++r) { for (int cc = 0; cc < 3; ++cc) dR[r*3+cc] = df[g*16 + r*4 + cc];
                                    at[g][r] = df[g*16 + r*4 + 3]; }
      mm3(dR, rot, aR[g]);
    }
    float R5[9], t5[3]; mm3(aR[4], aR[5], R5); mv3(aR[4], at[5], t5);
    for (int e = 0; e < 3; ++e) t5[e] += at[4][e];
    float R6[9], t6[3]; mm3(R5, aR[6], R6); mv3(R5, at[6], t6);
    for (int e = 0; e < 3; ++e) t6[e] += t5[e];
    float R7[9], t7[3]; mm3(R6, aR[7], R7); mv3(R6, at[7], t7);
    for (int e = 0; e < 3; ++e) t7[e] += t6[e];
    float allR[8][9], allt[8][3];
    for (int g = 0; g < 5; ++g) { for (int e = 0; e < 9; ++e) allR[g][e] = aR[g][e];
                                  for (int e = 0; e < 3; ++e) allt[g][e] = at[g][e]; }
    for (int e = 0; e < 9; ++e) { allR[5][e] = R5[e]; allR[6][e] = R6[e]; allR[7][e] = R7[e]; }
    for (int e = 0; e < 3; ++e) { allt[5][e] = t5[e]; allt[6][e] = t6[e]; allt[7][e] = t7[e]; }
    for (int g = 0; g < 8; ++g) {
      float tmp[9]; mm3(Rb[0], allR[g], tmp);
      float tv[3];  mv3(Rb[0], allt[g], tv);
      for (int e = 0; e < 9; ++e) gR[g][e] = tmp[e];
      for (int e = 0; e < 3; ++e) gt[g][e] = tv[e] + tb[0][e];
    }
  }
  for (int a = 0; a < NATOM; ++a) {
    int g = gidx[bu * NATOM + a];
    const float* lp = lit + ((size_t)bu * NATOM + a) * 3;
    float pv[3]; mv3(gR[g], lp, pv);
    float m = amask[bu * NATOM + a];
    #pragma unroll
    for (int d = 0; d < 3; ++d)
      out_pos[((size_t)n * NATOM + a) * 3 + d] = (pv[d] + gt[g][d]) * m;
  }
}

// ---------------- host orchestration ----------------
static inline void gemm(hipStream_t st, const float* in, int ldi, const float* W,
                        const float* bias, float* out, int ldo,
                        int M, int K, int Nc, int flags) {
  dim3 g((M + 63) / 64, (Nc + 127) / 128);
  k_gemm<<<g, 256, 0, st>>>(in, ldi, W, bias, out, ldo, M, K, Nc, flags);
}

extern "C" void kernel_launch(void* const* d_in, const int* in_sizes, int n_in,
                              void* d_out, int out_size, void* d_ws, size_t ws_size,
                              hipStream_t stream) {
  (void)in_sizes; (void)n_in; (void)out_size; (void)ws_size;
  const float* single  = (const float*)d_in[0];
  const float* pair    = (const float*)d_in[1];
  const int*   butype  = (const int*)d_in[2];
  const float* maskp   = (const float*)d_in[3];
  const int*   gidx    = (const int*)d_in[6];
  const float* ln_s_g  = (const float*)d_in[7];
  const float* ln_s_b  = (const float*)d_in[8];
  const float* ln_z_g  = (const float*)d_in[9];
  const float* ln_z_b  = (const float*)d_in[10];
  const float* in_w    = (const float*)d_in[11];
  const float* in_b    = (const float*)d_in[12];
  const float* q_w     = (const float*)d_in[13];
  const float* q_b     = (const float*)d_in[14];
  const float* kv_w    = (const float*)d_in[15];
  const float* kv_b    = (const float*)d_in[16];
  const float* qp_w    = (const float*)d_in[17];
  const float* qp_b    = (const float*)d_in[18];
  const float* kvp_w   = (const float*)d_in[19];
  const float* kvp_b   = (const float*)d_in[20];
  const float* b_w     = (const float*)d_in[21];
  const float* b_b     = (const float*)d_in[22];
  const float* gamma   = (const float*)d_in[23];
  const float* out_w   = (const float*)d_in[24];
  const float* out_b   = (const float*)d_in[25];
  const float* lnipa_g = (const float*)d_in[26];
  const float* lnipa_b = (const float*)d_in[27];
  const float* trln_g  = (const float*)d_in[28];
  const float* trln_b  = (const float*)d_in[29];
  const float* ref_w   = (const float*)d_in[30];
  const float* ref_b   = (const float*)d_in[31];
  const float* arin_w  = (const float*)d_in[32];
  const float* arin_b  = (const float*)d_in[33];
  const float* arini_w = (const float*)d_in[34];
  const float* arini_b = (const float*)d_in[35];
  const float* arout_w = (const float*)d_in[36];
  const float* arout_b = (const float*)d_in[37];
  const float* dframes = (const float*)d_in[38];
  const float* amask   = (const float*)d_in[39];
  const float* lit     = (const float*)d_in[40];
  const float* tr_w1   = (const float*)d_in[41];
  const float* tr_b1   = (const float*)d_in[42];
  const float* tr_w2   = (const float*)d_in[43];
  const float* tr_b2   = (const float*)d_in[44];
  const float* tr_w3   = (const float*)d_in[45];
  const float* tr_b3   = (const float*)d_in[46];
  const float* ar_w1[2] = { (const float*)d_in[47], (const float*)d_in[51] };
  const float* ar_b1[2] = { (const float*)d_in[48], (const float*)d_in[52] };
  const float* ar_w2[2] = { (const float*)d_in[49], (const float*)d_in[53] };
  const float* ar_b2[2] = { (const float*)d_in[50], (const float*)d_in[54] };

  float* ws = (float*)d_ws;
  float* s      = ws;                       // 512*384
  float* sinit  = s      + NRES * CS;
  float* xyz    = sinit  + NRES * CS;       // 512*24
  float* Rb     = xyz    + NRES * 24;       // 512*9
  float* tb     = Rb     + NRES * 9;        // 512*3
  float* qbuf   = tb     + NRES * 3;        // 512*192
  float* kvbuf  = qbuf   + NRES * 192;      // 512*384
  float* qpbuf  = kvbuf  + NRES * 384;      // 512*144
  float* qpa    = qpbuf  + NRES * 144;      // 512*144
  float* kvpbuf = qpa    + NRES * 144;      // 512*432
  float* kvpa   = kvpbuf + NRES * 432;      // 512*432
  float* biasbf = kvpa   + NRES * 432;      // 512*512*12
  _Float16* probsb = (_Float16*)(biasbf + (size_t)NRES * NRES * HNUM); // 512*512*12 f16
  float* catbuf = biasbf + (size_t)NRES * NRES * HNUM
                         + (size_t)NRES * NRES * HNUM / 2;             // 512*2112
  float* optraw = catbuf + (size_t)NRES * CATW;   // 512*288
  float* t1     = optraw + NRES * 288;
  float* t2     = t1     + NRES * CS;
  float* arx    = t2     + NRES * CS;
  float* arh    = arx    + NRES * CRES;
  float* arini  = arh    + NRES * CRES;
  float* unbuf  = arini  + NRES * CRES;
  float* angbuf = unbuf  + NRES * NA * 2;

  float* dout = (float*)d_out;
  const size_t OFF_POS = 0;
  const size_t OFF_ANG = (size_t)NBLK * NRES * NATOM * 3;
  const size_t OFF_UN  = OFF_ANG + (size_t)NBLK * NRES * NA * 2;
  const size_t OFF_S   = OFF_UN  + (size_t)NBLK * NRES * NA * 2;

  k_ln<<<NRES, 128, 0, stream>>>(single, ln_s_g, ln_s_b, sinit, CS);
  k_pair_bias<<<NRES * NRES / 8, 256, 0, stream>>>(pair, ln_z_g, ln_z_b, b_w, b_b, biasbf);
  gemm(stream, sinit, CS, in_w, in_b, s, CS, NRES, CS, CS, 0);
  gemm(stream, sinit, CS, arini_w, arini_b, arini, CRES, NRES, CS, CRES, GEMM_RELU_IN);
  k_zero<<<(NRES * 24 + 255) / 256, 256, 0, stream>>>(xyz, NRES * 24);

  for (int bi = 0; bi < NBLK; ++bi) {
    // ---- IPA ----
    k_frames<<<2, 256, 0, stream>>>(xyz, Rb, tb);
    gemm(stream, s, CS, q_w,   q_b,   qbuf,   192, NRES, CS, 192, 0);
    gemm(stream, s, CS, kv_w,  kv_b,  kvbuf,  384, NRES, CS, 384, 0);
    gemm(stream, s, CS, qp_w,  qp_b,  qpbuf,  144, NRES, CS, 144, 0);
    gemm(stream, s, CS, kvp_w, kvp_b, kvpbuf, 432, NRES, CS, 432, 0);
    k_apply<<<(NRES * 48 + 255) / 256, 256, 0, stream>>>(qpbuf, Rb, tb, qpa, 48);
    k_apply<<<(NRES * 144 + 255) / 256, 256, 0, stream>>>(kvpbuf, Rb, tb, kvpa, 144);
    k_attn1<<<dim3(NRES / 16, HNUM), 256, 0, stream>>>(qbuf, kvbuf, qpa, kvpa, biasbf,
                                                       maskp, gamma, probsb, catbuf,
                                                       optraw);
    k_pair<<<NRES, 256, 0, stream>>>(probsb, pair, ln_z_g, ln_z_b, catbuf);
    k_finalize<<<(NRES * 96 + 255) / 256, 256, 0, stream>>>(optraw, Rb, tb, catbuf);
    gemm(stream, catbuf, CATW, out_w, out_b, s, CS, NRES, CATW, CS, GEMM_ACCUM);
    k_ln<<<NRES, 128, 0, stream>>>(s, lnipa_g, lnipa_b, s, CS);

    // ---- transition ----
    gemm(stream, s,  CS, tr_w1, tr_b1, t1, CS, NRES, CS, CS, 0);
    gemm(stream, t1, CS, tr_w2, tr_b2, t2, CS, NRES, CS, CS, GEMM_RELU_IN);
    gemm(stream, t2, CS, tr_w3, tr_b3, s,  CS, NRES, CS, CS, GEMM_RELU_IN | GEMM_ACCUM);
    k_ln<<<NRES, 128, 0, stream>>>(s, trln_g, trln_b, s, CS);

    // ---- backbone refine ----
    gemm(stream, s, CS, ref_w, ref_b, xyz, 24, NRES, CS, 24, GEMM_ACCUM);

    // ---- angle resnet ----
    k_copy<<<(NRES * CRES + 255) / 256, 256, 0, stream>>>(arini, arx, NRES * CRES);
    gemm(stream, s, CS, arin_w, arin_b, arx, CRES, NRES, CS, CRES,
         GEMM_RELU_IN | GEMM_ACCUM);
    for (int rb = 0; rb < 2; ++rb) {
      gemm(stream, arx, CRES, ar_w1[rb], ar_b1[rb], arh, CRES, NRES, CRES, CRES,
           GEMM_RELU_IN);
      gemm(stream, arh, CRES, ar_w2[rb], ar_b2[rb], arx, CRES, NRES, CRES, CRES,
           GEMM_RELU_IN | GEMM_ACCUM);
    }
    gemm(stream, arx, CRES, arout_w, arout_b, unbuf, NA * 2, NRES, CRES, NA * 2,
         GEMM_RELU_IN);
    k_angles<<<(NRES * NA + 255) / 256, 256, 0, stream>>>(
        unbuf, angbuf,
        dout + OFF_UN  + (size_t)bi * NRES * NA * 2,
        dout + OFF_ANG + (size_t)bi * NRES * NA * 2);

    // ---- torsion frames + atoms ----
    k_atoms<<<2, 256, 0, stream>>>(xyz, butype, angbuf, dframes, gidx, lit, amask,
                                   dout + OFF_POS + (size_t)bi * NRES * NATOM * 3);
  }
  hipMemcpyAsync(dout + OFF_S, s, (size_t)NRES * CS * sizeof(float),
                 hipMemcpyDeviceToDevice, stream);
}
